// Attention_Layer_56281251447188
// MI455X (gfx1250) — matrix operand order repaired and checked by reading, not compiled, not run on hardware
//
#include <hip/hip_runtime.h>
#include <hip/hip_bf16.h>
#include <cstddef>

// ---------------------------------------------------------------------------
// B=16384, N=32, D_IN=D_H=D_OUT=128, H=4.
// Algebra: logits = (x^T Wq^T Wk) nbr  -> precompute A_h = Wq_h^T Wk_h.
//          res    = (1/H) sum_h Wv_h (nbr^T attn_h) -> weighted-sum u first,
//          then one GEMM vs Wv_cat. Removes the big K/V projections entirely.
// Workspace layout (f16, needs 288KB of d_ws):
//   A2t  [512][128]  A2t[h*128+j][i]   = sum_d Wq[h][d][i]*Wk[h][d][j]
//   Wv2t [128][512]  Wv2t[o][h*128+j]  = 0.25f * Wv[h][o][j]
//   Wo2t [128][128]  Wo2t[o][d]        = Wo[o][d]
// All stored [col][K] so WMMA B-fragments are contiguous per lane.
// ---------------------------------------------------------------------------

typedef __attribute__((ext_vector_type(16))) _Float16 v16h;
typedef __attribute__((ext_vector_type(8)))  _Float16 v8h;
typedef __attribute__((ext_vector_type(8)))  float    v8f;

#define TB 16            // batch rows per workgroup
#define XS 136           // sX row stride (halves), 16B aligned, bank-skewed
#define SS 520           // sS/sU row stride (halves)
#define RS 136           // sR row stride (halves)

static __device__ __forceinline__ v8f wmma16(v16h a, v16h b, v8f c) {
  // (neg_a, A, neg_b, B, c_mod, C, reuse_a, reuse_b)
  return __builtin_amdgcn_wmma_f32_16x16x32_f16(false, a, false, b,
                                                (short)0, c, false, false);
}

// A-fragment (16x32 f16) from row-major f16 LDS tile.
// Lane layout: M = lane&15; lanes 0-15 hold K = kb+[0..15];
// lanes 16-31 hold K = kb+[16..31].
static __device__ __forceinline__ v16h load_a_lds(const _Float16* base, int stride,
                                                  int kb, int lane) {
  const int m  = lane & 15;
  const int h8 = lane >> 4;
  const _Float16* row = base + m * stride + kb + h8 * 16;
  v8h lo = *(const v8h*)(row);
  v8h hi = *(const v8h*)(row + 8);
  return __builtin_shufflevector(lo, hi, 0,1,2,3,4,5,6,7,8,9,10,11,12,13,14,15);
}

// B-fragment (32x16 f16) from global matrix stored [col][K] (krows = K extent).
// Lane holds 16 consecutive K of column (lane&15), K base = kb + 16*(lane>>4).
static __device__ __forceinline__ v16h load_b_glb(const _Float16* __restrict__ bt,
                                                  int krows, int colBase,
                                                  int kb, int lane) {
  const int n = lane & 15;
  const int k = kb + ((lane >> 4) << 4);
  return *(const v16h*)(bt + (size_t)(colBase + n) * krows + k);
}

// C/D 16x16 f32: VGPR r -> M = r + 8*(lane>>4), N = lane&15.
static __device__ __forceinline__ void store_c_f16(_Float16* dst, int stride,
                                                   int colBase, int lane, v8f c) {
  const int n  = lane & 15;
  const int mh = (lane >> 4) * 8;
#pragma unroll
  for (int r = 0; r < 8; ++r)
    dst[(mh + r) * stride + colBase + n] = (_Float16)c[r];
}

// ---------------------------------------------------------------------------
// Prep kernel: fold weights into f16 B-operand matrices (one-time, tiny).
// ---------------------------------------------------------------------------
__global__ __launch_bounds__(256) void attn_prep(const float* __restrict__ Wq,
                                                 const float* __restrict__ Wk,
                                                 const float* __restrict__ Wv,
                                                 const float* __restrict__ Wo,
                                                 _Float16* __restrict__ A2t,
                                                 _Float16* __restrict__ Wv2t,
                                                 _Float16* __restrict__ Wo2t) {
  const int tid = blockIdx.x * 256 + threadIdx.x;
  if (tid < 512 * 128) {                       // A2t[col][i], col = h*128+j
    const int i = tid & 127, col = tid >> 7;
    const int h = col >> 7, j = col & 127;
    const float* wq = Wq + h * 128 * 128;
    const float* wk = Wk + h * 128 * 128;
    float acc = 0.f;
#pragma unroll 8
    for (int d = 0; d < 128; ++d)
      acc += wq[d * 128 + i] * wk[d * 128 + j];
    A2t[(size_t)col * 128 + i] = (_Float16)acc;
  } else if (tid < 2 * 512 * 128) {            // Wv2t[o][k], k = h*128+j
    const int t = tid - 512 * 128;
    const int k = t & 511, o = t >> 9;
    const int h = k >> 7, j = k & 127;
    Wv2t[(size_t)o * 512 + k] = (_Float16)(0.25f * Wv[(h * 128 + o) * 128 + j]);
  } else if (tid < 2 * 512 * 128 + 128 * 128) { // Wo2t[o][d] = Wo row-major
    const int t = tid - 2 * 512 * 128;
    Wo2t[t] = (_Float16)Wo[t];
  }
}

// ---------------------------------------------------------------------------
// Main kernel: one WG (8 wave32) per 16 batch rows.
// ---------------------------------------------------------------------------
__global__ __launch_bounds__(256) void attn_main(const float* __restrict__ x,
                                                 const float* __restrict__ nbr,
                                                 const float* __restrict__ bo,
                                                 const _Float16* __restrict__ A2t,
                                                 const _Float16* __restrict__ Wv2t,
                                                 const _Float16* __restrict__ Wo2t,
                                                 float* __restrict__ out) {
  __shared__ __align__(16) _Float16 sX[TB * XS];   // x tile, f16
  __shared__ __align__(16) _Float16 sS[TB * SS];   // s = x^T A, f16
  __shared__ __align__(16) _Float16 sU[TB * SS];   // u = attn-weighted nbrs, f16
  __shared__ __align__(16) _Float16 sR[TB * RS];   // res (pre-Wo), f16
  __shared__ float sAttn[TB * 4 * 32];             // softmax weights

  const int tid  = threadIdx.x;
  const int lane = tid & 31;
  const int wv   = tid >> 5;                       // 8 waves
  const int b0   = blockIdx.x * TB;

  // ---- Phase L: stage x tile as f16 -------------------------------------
#pragma unroll
  for (int p = 0; p < (TB * 128) / 256; ++p) {
    const int idx = tid + p * 256;
    const int bi = idx >> 7, i = idx & 127;
    sX[bi * XS + i] = (_Float16)x[(size_t)(b0 + bi) * 128 + i];
  }
  __syncthreads();

  // ---- Phase 1: S[16x512] = X[16x128] @ A2 (WMMA, 4 col-tiles per wave) --
  {
    v16h a[4];
#pragma unroll
    for (int ks = 0; ks < 4; ++ks) a[ks] = load_a_lds(sX, XS, ks * 32, lane);
#pragma unroll
    for (int t = 0; t < 4; ++t) {
      const int colBase = (wv * 4 + t) * 16;
      v8f acc = {};
#pragma unroll
      for (int ks = 0; ks < 4; ++ks) {
        v16h b = load_b_glb(A2t, 128, colBase, ks * 32, lane);
        acc = wmma16(a[ks], b, acc);
      }
      store_c_f16(sS, SS, colBase, lane, acc);
    }
  }
  __syncthreads();

  // ---- Phase 2: logits + softmax + u (VALU; wave handles 2 batches) ------
#pragma unroll
  for (int bb = 0; bb < 2; ++bb) {
    const int bt = wv * 2 + bb;
    const float* nrow = nbr + ((size_t)(b0 + bt) * 32 + lane) * 128; // lane = neighbor
    // logits[h][lane] = s_h . nbr_lane  (f32 accumulate)
#pragma unroll
    for (int h = 0; h < 4; ++h) {
      const _Float16* srow = sS + bt * SS + h * 128;   // broadcast reads
      float acc = 0.f;
#pragma unroll 8
      for (int i = 0; i < 128; i += 4) {
        float4 nv = *(const float4*)(nrow + i);
        acc += nv.x * (float)srow[i]     + nv.y * (float)srow[i + 1]
             + nv.z * (float)srow[i + 2] + nv.w * (float)srow[i + 3];
      }
      // softmax across the 32 neighbors (= across lanes, wave32 butterfly)
      float m = acc;
#pragma unroll
      for (int off = 16; off > 0; off >>= 1) m = fmaxf(m, __shfl_xor(m, off, 32));
      const float e = __expf(acc - m);
      float ssum = e;
#pragma unroll
      for (int off = 16; off > 0; off >>= 1) ssum += __shfl_xor(ssum, off, 32);
      sAttn[(bt * 4 + h) * 32 + lane] = e / ssum;
    }
    // u[h][j] = sum_n attn[h][n] * nbr[n][j]; lane covers j = 4*lane..4*lane+3
    const float* nbase = nbr + (size_t)(b0 + bt) * 4096 + lane * 4;
#pragma unroll
    for (int h = 0; h < 4; ++h) {
      const float* aw = &sAttn[(bt * 4 + h) * 32];
      float u0 = 0.f, u1 = 0.f, u2 = 0.f, u3 = 0.f;
#pragma unroll 8
      for (int n = 0; n < 32; ++n) {
        const float w = aw[n];                     // LDS broadcast
        float4 nv = *(const float4*)(nbase + n * 128);  // coalesced, L2-hot
        u0 += w * nv.x; u1 += w * nv.y; u2 += w * nv.z; u3 += w * nv.w;
      }
      _Float16* up = sU + bt * SS + h * 128 + lane * 4;
      up[0] = (_Float16)u0; up[1] = (_Float16)u1;
      up[2] = (_Float16)u2; up[3] = (_Float16)u3;
    }
  }
  __syncthreads();

  // ---- Phase 4: R[16x128] = U[16x512] @ Wv2t (WMMA, 1 col-tile per wave) -
  {
    const int colBase = wv * 16;
    v8f acc = {};
#pragma unroll
    for (int ks = 0; ks < 16; ++ks) {
      v16h a = load_a_lds(sU, SS, ks * 32, lane);
      v16h b = load_b_glb(Wv2t, 512, colBase, ks * 32, lane);
      acc = wmma16(a, b, acc);
    }
    store_c_f16(sR, RS, colBase, lane, acc);
  }
  __syncthreads();

  // ---- Phase 5: O = R @ Wo2t + bo, leaky-relu, store f32 -----------------
  {
    const int colBase = wv * 16;
    v8f acc = {};
#pragma unroll
    for (int ks = 0; ks < 4; ++ks) {
      v16h a = load_a_lds(sR, RS, ks * 32, lane);
      v16h b = load_b_glb(Wo2t, 128, colBase, ks * 32, lane);
      acc = wmma16(a, b, acc);
    }
    const int n = lane & 15;
    const int col = colBase + n;
    const float bias = bo[col];
    const int mh = (lane >> 4) * 8;
#pragma unroll
    for (int r = 0; r < 8; ++r) {
      float v = acc[r] + bias;
      v = v > 0.f ? v : 0.01f * v;
      out[(size_t)(b0 + mh + r) * 128 + col] = v;
    }
  }
}

// ---------------------------------------------------------------------------
extern "C" void kernel_launch(void* const* d_in, const int* in_sizes, int n_in,
                              void* d_out, int out_size, void* d_ws, size_t ws_size,
                              hipStream_t stream) {
  const float* x   = (const float*)d_in[0];
  const float* nbr = (const float*)d_in[1];
  const float* Wq  = (const float*)d_in[2];
  const float* Wk  = (const float*)d_in[3];
  const float* Wv  = (const float*)d_in[4];
  const float* Wo  = (const float*)d_in[5];
  const float* bo  = (const float*)d_in[6];
  float* out = (float*)d_out;

  _Float16* A2t  = (_Float16*)d_ws;          // 512*128 f16 = 128KB
  _Float16* Wv2t = A2t + 512 * 128;          // 128*512 f16 = 128KB
  _Float16* Wo2t = Wv2t + 128 * 512;         // 128*128 f16 = 32KB  (288KB total)

  const int B = in_sizes[0] / 128;           // 16384

  attn_prep<<<(2 * 512 * 128 + 128 * 128 + 255) / 256, 256, 0, stream>>>(
      Wq, Wk, Wv, Wo, A2t, Wv2t, Wo2t);
  attn_main<<<B / TB, 256, 0, stream>>>(x, nbr, bo, A2t, Wv2t, Wo2t, out);
}
